// HardActor_31937376813217
// MI455X (gfx1250) — matrix operand order repaired and checked by reading, not compiled, not run on hardware
//
#include <hip/hip_runtime.h>
#include <math.h>

typedef __attribute__((ext_vector_type(16))) _Float16 v16h;
typedef __attribute__((ext_vector_type(8)))  _Float16 v8h;
typedef __attribute__((ext_vector_type(8)))  float    v8f;
typedef __attribute__((ext_vector_type(4)))  int      v4i;

#define IN_DIM   64
#define HIDDEN   256
#define N_ASSETS 32
#define MT       8            // 16-row M-tiles per block -> 128 samples/block
#define ROWS     (MT * 16)
#define XLDh     72           // 64+8 halfs; stride 36 dwords -> conflict-free row spread
#define HLDh     264          // 256+8 halfs; stride 132 dwords -> conflict-free

// ws layout (halfs): W1t[256][64] @0, W2t[256][256] @16384, HWt[128][256] @81920
#define W1T_OFF  0
#define W2T_OFF  16384
#define HWT_OFF  81920
#define PREP_N   114688

__device__ __forceinline__ v8f wmma16(v16h a, v16h b, v8f c) {
  // D(16x16 f32) = A(16x32 f16) * B(32x16 f16) + C, fp32 accumulate
  return __builtin_amdgcn_wmma_f32_16x16x32_f16(
      false, a, false, b, (short)0, c, false, false);
}

// One MMA pass: 16 output columns (this wave's tile), MT row-tiles, K = KDIM.
// aBase = activations + mrow*ALD + 16*kh (so all A offsets are constants)
// wcol  = weight column + 16*kh          (so all B offsets are constants)
template<int KDIM, int ALD>
__device__ __forceinline__ void mma_pass(const _Float16* __restrict__ aBase,
                                         const _Float16* __restrict__ wcol,
                                         v8f acc[MT]) {
#pragma unroll
  for (int m = 0; m < MT; ++m) acc[m] = (v8f){};
#pragma unroll 2
  for (int k0 = 0; k0 < KDIM; k0 += 32) {
    const v16h b = *(const v16h*)(wcol + k0);        // K = k0+16*kh .. +15
#pragma unroll
    for (int m = 0; m < MT; ++m) {
      union { v16h v; v8h h[2]; } a;
      a.h[0] = *(const v8h*)(aBase + m * 16 * ALD + k0);       // K = k0+16*kh..+7
      a.h[1] = *(const v8h*)(aBase + m * 16 * ALD + k0 + 8);   // +8
      acc[m] = wmma16(a.v, b, acc[m]);
    }
  }
}

// bias + relu + f16 store of a C/D tile set; outBase = dst + 8*kh*HLDh + col
__device__ __forceinline__ void store_act(_Float16* __restrict__ outBase,
                                          const v8f acc[MT], float bias) {
#pragma unroll
  for (int m = 0; m < MT; ++m)
#pragma unroll
    for (int v = 0; v < 8; ++v)
      outBase[(m * 16 + v) * HLDh] = (_Float16)fmaxf(acc[m][v] + bias, 0.0f);
}

// ---- one-shot weight convert + transpose into workspace (f16, column-major) ----
__global__ __launch_bounds__(256) void prep_kernel(
    const float* __restrict__ W1, const float* __restrict__ W2,
    const float* __restrict__ hW, _Float16* __restrict__ ws)
{
  const int i = blockIdx.x * 256 + threadIdx.x;
  if (i < W2T_OFF) {                       // W1t[n*64+k] = W1[k*256+n]
    const int n = i >> 6, k = i & 63;
    ws[i] = (_Float16)W1[k * HIDDEN + n];
  } else if (i < HWT_OFF) {                // W2t[n*256+k] = W2[k*256+n]
    const int j = i - W2T_OFF;
    const int n = j >> 8, k = j & 255;
    ws[i] = (_Float16)W2[k * HIDDEN + n];
  } else if (i < PREP_N) {                 // HWt[c*256+k] = heads_W[r][k][a], c=r*32+a
    const int j = i - HWT_OFF;
    const int c = j >> 8, k = j & 255;
    const int r = c >> 5, a = c & 31;
    ws[i] = (_Float16)hW[(r * HIDDEN + k) * N_ASSETS + a];
  }
}

__global__ __launch_bounds__(256, 4) void actor_kernel(
    const float* __restrict__ x,
    const float* __restrict__ b1, const float* __restrict__ b2,
    const float* __restrict__ hb, const float* __restrict__ log_std,
    const _Float16* __restrict__ W1t, const _Float16* __restrict__ W2t,
    const _Float16* __restrict__ HWt,
    float* __restrict__ out_mean, float* __restrict__ out_std)
{
  __shared__ _Float16 sXh[ROWS * XLDh];   // x tile, f16       (~18 KB)
  __shared__ _Float16 sH [ROWS * HLDh];   // relu layer-1, f16 (~66 KB)
  __shared__ _Float16 sF [ROWS * HLDh];   // relu layer-2, f16 (~66 KB)
  __shared__ int      sReg[ROWS];

  const int tid  = threadIdx.x;
  const int wave = tid >> 5;
  const int lane = tid & 31;
  const int mrow = lane & 15;            // M (A-frag) / N (B-frag) within tile
  const int kh   = lane >> 4;            // K-half select
  const long long base = (long long)blockIdx.x * ROWS;

  // ---- stage x tile (f32 -> f16, non-temporal: streamed once) + regime ids
  for (int i = tid; i < ROWS * IN_DIM; i += 256) {
    const int m = i >> 6, k = i & 63;
    sXh[m * XLDh + k] = (_Float16)__builtin_nontemporal_load(&x[(base + m) * IN_DIM + k]);
  }
  if (tid < ROWS)
    sReg[tid] = (int)__builtin_nontemporal_load(&x[(base + tid) * IN_DIM + (IN_DIM - 1)]);
  __syncthreads();

  // ---- layer 1: h = relu(x @ W1 + b1); wave owns 2 of 16 N-tiles,
  //      B-frag registered and reused across all 8 M-tiles.
#pragma unroll
  for (int t = 0; t < 2; ++t) {
    const int col = (wave * 2 + t) * 16 + mrow;
    v8f acc[MT];
    mma_pass<IN_DIM, XLDh>(sXh + mrow * XLDh + 16 * kh,
                           W1t + (size_t)col * IN_DIM + 16 * kh, acc);
    store_act(sH + (8 * kh) * HLDh + col, acc, b1[col]);
  }
  __syncthreads();

  // ---- layer 2: f = relu(h @ W2 + b2)
#pragma unroll
  for (int t = 0; t < 2; ++t) {
    const int col = (wave * 2 + t) * 16 + mrow;
    v8f acc[MT];
    mma_pass<HIDDEN, HLDh>(sH + mrow * HLDh + 16 * kh,
                           W2t + (size_t)col * HIDDEN + 16 * kh, acc);
    store_act(sF + (8 * kh) * HLDh + col, acc, b2[col]);
  }
  __syncthreads();

  // ---- regime heads: combined [HIDDEN x 128] (4 regimes x 32 assets);
  //      wave w owns columns [w*16, w*16+16) -> regime r = w/2 wave-uniform,
  //      per-sample dispatch = predicated non-temporal store.
  {
    const int c = wave * 16 + mrow;
    const int r = c >> 5;
    const int aidx = c & 31;
    v8f acc[MT];
    mma_pass<HIDDEN, HLDh>(sF + mrow * HLDh + 16 * kh,
                           HWt + (size_t)c * HIDDEN + 16 * kh, acc);
    const float bias = hb[r * N_ASSETS + aidx];
#pragma unroll
    for (int m = 0; m < MT; ++m) {
      union { v4i v[2]; int s[8]; } reg8;
      reg8.v[0] = *(const v4i*)(sReg + m * 16 + 8 * kh);      // rows +0..3
      reg8.v[1] = *(const v4i*)(sReg + m * 16 + 8 * kh + 4);  // rows +4..7
      float* op = out_mean + (base + m * 16 + 8 * kh) * N_ASSETS + aidx;
#pragma unroll
      for (int v = 0; v < 8; ++v)
        if (reg8.s[v] == r)
          __builtin_nontemporal_store((acc[m][v] + bias) * 0.1f, op + v * N_ASSETS);
    }
  }

  // ---- std = clip(exp(log_std), 1e-3, 1) broadcast per sample
  for (int i = tid; i < ROWS * N_ASSETS; i += 256) {
    const int m = i >> 5, a = i & 31;
    float s = expf(log_std[a]);
    s = fminf(fmaxf(s, 1e-3f), 1.0f);
    __builtin_nontemporal_store(s, &out_std[(base + m) * N_ASSETS + a]);
  }
}

extern "C" void kernel_launch(void* const* d_in, const int* in_sizes, int n_in,
                              void* d_out, int out_size, void* d_ws, size_t ws_size,
                              hipStream_t stream) {
  const float* x       = (const float*)d_in[0];
  const float* W1      = (const float*)d_in[1];
  const float* b1      = (const float*)d_in[2];
  const float* W2      = (const float*)d_in[3];
  const float* b2      = (const float*)d_in[4];
  const float* hW      = (const float*)d_in[5];
  const float* hb      = (const float*)d_in[6];
  const float* log_std = (const float*)d_in[7];

  const int B = in_sizes[0] / IN_DIM;                 // 262144
  float* out_mean = (float*)d_out;                    // [B, 32]
  float* out_std  = out_mean + (size_t)B * N_ASSETS;  // [B, 32]

  _Float16* wsh = (_Float16*)d_ws;                    // 229 KB used

  hipLaunchKernelGGL(prep_kernel, dim3((PREP_N + 255) / 256), dim3(256), 0, stream,
                     W1, W2, hW, wsh);
  hipLaunchKernelGGL(actor_kernel, dim3(B / ROWS), dim3(256), 0, stream,
                     x, b1, b2, hb, log_std,
                     wsh + W1T_OFF, wsh + W2T_OFF, wsh + HWT_OFF,
                     out_mean, out_std);
}
